// TransformerEncoderLayer_19550691131964
// MI455X (gfx1250) — matrix operand order repaired and checked by reading, not compiled, not run on hardware
//
#include <hip/hip_runtime.h>

typedef __attribute__((ext_vector_type(16))) _Float16 v16h;
typedef __attribute__((ext_vector_type(8)))  float    v8f;
typedef __attribute__((ext_vector_type(4)))  int      v4i;

union HFrag { v16h h; v4i i[2]; };

#define D_MODEL  1024
#define N_HEADS  16
#define HEAD_DIM 64
#define D_FF     4096
#define SEQ      2048
#define BATCH    4
#define ROWS     (BATCH * SEQ)   // 8192 token rows

// ---------------------------------------------------------------------------
// Weight prep: W [K x N] f32 (row-major)  ->  WT [N x K] f16 (row-major)
// so WMMA B-fragments become contiguous 32B loads per lane.
// ---------------------------------------------------------------------------
__global__ void transpose_to_f16(const float* __restrict__ W,
                                 _Float16* __restrict__ WT, int K, int N) {
  __shared__ float tile[32][33];
  int kt = blockIdx.x * 32;
  int nt = blockIdx.y * 32;
  int tx = threadIdx.x & 31;
  int ty = threadIdx.x >> 5;  // 0..7
#pragma unroll
  for (int i = 0; i < 4; ++i)
    tile[ty + 8 * i][tx] = W[(size_t)(kt + ty + 8 * i) * N + nt + tx];
  __syncthreads();
#pragma unroll
  for (int i = 0; i < 4; ++i)
    WT[(size_t)(nt + ty + 8 * i) * K + kt + tx] = (_Float16)tile[tx][ty + 8 * i];
}

// ---------------------------------------------------------------------------
// Mean-only layernorm: out = w*(x-mean) + b  (f16 output for WMMA A operand)
// ---------------------------------------------------------------------------
__global__ void ln_novar_f16(const float* __restrict__ X, const float* __restrict__ w,
                             const float* __restrict__ b, _Float16* __restrict__ outh) {
  __shared__ float red[256];
  int row = blockIdx.x;
  const float* x = X + (size_t)row * D_MODEL;
  float s = 0.f;
  for (int c = threadIdx.x; c < D_MODEL; c += 256) s += x[c];
  red[threadIdx.x] = s;
  __syncthreads();
  for (int o = 128; o > 0; o >>= 1) {
    if (threadIdx.x < o) red[threadIdx.x] += red[threadIdx.x + o];
    __syncthreads();
  }
  float u = red[0] * (1.0f / D_MODEL);
  for (int c = threadIdx.x; c < D_MODEL; c += 256)
    outh[(size_t)row * D_MODEL + c] = (_Float16)(w[c] * (x[c] - u) + b[c]);
}

// Same, but keeps an f32 copy for the final residual add.
__global__ void ln_novar_dual(const float* __restrict__ X, const float* __restrict__ w,
                              const float* __restrict__ b, float* __restrict__ outf,
                              _Float16* __restrict__ outh) {
  __shared__ float red[256];
  int row = blockIdx.x;
  const float* x = X + (size_t)row * D_MODEL;
  float s = 0.f;
  for (int c = threadIdx.x; c < D_MODEL; c += 256) s += x[c];
  red[threadIdx.x] = s;
  __syncthreads();
  for (int o = 128; o > 0; o >>= 1) {
    if (threadIdx.x < o) red[threadIdx.x] += red[threadIdx.x + o];
    __syncthreads();
  }
  float u = red[0] * (1.0f / D_MODEL);
  for (int c = threadIdx.x; c < D_MODEL; c += 256) {
    float v = w[c] * (x[c] - u) + b[c];
    outf[(size_t)row * D_MODEL + c] = v;
    outh[(size_t)row * D_MODEL + c] = (_Float16)v;
  }
}

// ---------------------------------------------------------------------------
// WMMA GEMM: C = A[MxK] * BT[NxK]^T  (f16 in, f32 accum)
// Block = 8 waves; each wave computes a 32x64 tile (2 A-frags x 4 B-frags ->
// 8 WMMAs per 32-K step). Double-buffered fragment loads hide global latency.
// Block tile = 256 rows x 64 cols. Requires K % 64 == 0 (true: 64/1024/4096).
// ---------------------------------------------------------------------------
#define EPI_QK          0  // f16 store into [B,H,seq,64]
#define EPI_VT          1  // f16 store into [B,H,64,seq] (transposed V)
#define EPI_H_BIAS_RELU 2  // f16 store, +bias, relu
#define EPI_F_BIAS_RES  3  // f32 store, +bias, +residual

struct KStep {
  HFrag a[2];
  HFrag b[4];
};

__device__ __forceinline__ void load_kstep(KStep& f, const _Float16* arow0,
                                           const _Float16* arow1,
                                           const _Float16* const bp[4],
                                           int k0, int hi) {
  const _Float16* p0 = arow0 + k0 + hi * 16;
  f.a[0].i[0] = *(const v4i*)p0;
  f.a[0].i[1] = *(const v4i*)(p0 + 8);
  const _Float16* p1 = arow1 + k0 + hi * 16;
  f.a[1].i[0] = *(const v4i*)p1;
  f.a[1].i[1] = *(const v4i*)(p1 + 8);
#pragma unroll
  for (int nt = 0; nt < 4; ++nt) {
    const _Float16* q = bp[nt] + k0;
    f.b[nt].i[0] = *(const v4i*)q;
    f.b[nt].i[1] = *(const v4i*)(q + 8);
  }
}

__device__ __forceinline__ void wmma8(const KStep& f, v8f acc[2][4]) {
#pragma unroll
  for (int rt = 0; rt < 2; ++rt)
#pragma unroll
    for (int nt = 0; nt < 4; ++nt)
      acc[rt][nt] = __builtin_amdgcn_wmma_f32_16x16x32_f16(
          false, f.a[rt].h, false, f.b[nt].h, (short)0, acc[rt][nt], false, false);
}

__global__ void __launch_bounds__(256)
gemm_wmma_f16(const _Float16* __restrict__ A, const _Float16* __restrict__ BT,
              const float* __restrict__ bias, const float* __restrict__ res,
              void* __restrict__ out, int M, int N, int K, int mode) {
  const int wv   = threadIdx.x >> 5;
  const int lane = threadIdx.x & 31;
  const int rlo  = lane & 15;
  const int hi   = lane >> 4;
  const int rowStart = blockIdx.x * 256 + wv * 32;
  const int colStart = blockIdx.y * 64;

  v8f acc[2][4] = {};
  const _Float16* arow0 = A + (size_t)(rowStart + rlo) * K;
  const _Float16* arow1 = arow0 + (size_t)16 * K;
  const _Float16* bp[4];
#pragma unroll
  for (int nt = 0; nt < 4; ++nt)
    bp[nt] = BT + (size_t)(colStart + nt * 16 + rlo) * K + hi * 16;

  KStep f0, f1;
  load_kstep(f0, arow0, arow1, bp, 0, hi);
  for (int k0 = 0; k0 < K; k0 += 64) {
    // prefetch the A stream two steps ahead
    if (k0 + 128 < K) __builtin_prefetch(arow0 + k0 + 128, 0, 1);
    load_kstep(f1, arow0, arow1, bp, k0 + 32, hi);   // next 32-K step (in flight)
    wmma8(f0, acc);                                   // consume current
    if (k0 + 64 < K) load_kstep(f0, arow0, arow1, bp, k0 + 64, hi);
    wmma8(f1, acc);
  }

#pragma unroll
  for (int rt = 0; rt < 2; ++rt) {
#pragma unroll
    for (int nt = 0; nt < 4; ++nt) {
#pragma unroll
      for (int j = 0; j < 8; ++j) {
        int row = rowStart + rt * 16 + j + 8 * hi;  // C layout: vgpr j -> row j+8*hi
        int col = colStart + nt * 16 + rlo;         // col = lane%16
        float v = acc[rt][nt][j];
        if (mode == EPI_QK) {
          int bb = row >> 11, tok = row & 2047;
          int hh = col >> 6,  dd  = col & 63;
          ((_Float16*)out)[((size_t)(bb * N_HEADS + hh) * SEQ + tok) * HEAD_DIM + dd] =
              (_Float16)v;
        } else if (mode == EPI_VT) {
          int bb = row >> 11, tok = row & 2047;
          int hh = col >> 6,  dd  = col & 63;
          ((_Float16*)out)[((size_t)(bb * N_HEADS + hh) * HEAD_DIM + dd) * SEQ + tok] =
              (_Float16)v;
        } else if (mode == EPI_H_BIAS_RELU) {
          float t = v + bias[col];
          t = t > 0.f ? t : 0.f;
          ((_Float16*)out)[(size_t)row * N + col] = (_Float16)t;
        } else {  // EPI_F_BIAS_RES
          ((float*)out)[(size_t)row * N + col] =
              v + bias[col] + res[(size_t)row * N + col];
        }
      }
    }
  }
}

// ---------------------------------------------------------------------------
// Fused attention: one block per (b, h, 16-row q block).
// 8 waves each own a 256-column strip of the 2048-wide score row.
// Scores stay register-resident (16 WMMA accumulators / wave); softmax stats
// reduced via shfl_xor + LDS; P transposed through per-wave LDS strip.
// ---------------------------------------------------------------------------
__global__ void __launch_bounds__(256)
attention_wmma(const _Float16* __restrict__ Q, const _Float16* __restrict__ Kf,
               const _Float16* __restrict__ VT, _Float16* __restrict__ O) {
  __shared__ _Float16 ldsP[8][16][32];  // per-wave P transpose strip (8 KB)
  __shared__ float    ldsO[8][16][64];  // per-wave O partials (32 KB)
  __shared__ float    ldsStat[8][16];
  __shared__ float    ldsM[16];
  __shared__ float    ldsL[16];

  const int wv   = threadIdx.x >> 5;
  const int lane = threadIdx.x & 31;
  const int rlo  = lane & 15;
  const int hi   = lane >> 4;

  const int bh   = blockIdx.x >> 7;   // b*16 + h
  const int qblk = blockIdx.x & 127;
  const _Float16* qbase = Q  + ((size_t)bh * SEQ + qblk * 16) * HEAD_DIM;
  const _Float16* kbase = Kf + (size_t)bh * SEQ * HEAD_DIM;
  const _Float16* vbase = VT + (size_t)bh * HEAD_DIM * SEQ;

  // Q block A-fragments (K=64 -> two K32 fragments), reused for all 16 tiles
  HFrag aq[2];
#pragma unroll
  for (int kt = 0; kt < 2; ++kt) {
    const _Float16* p = qbase + (size_t)rlo * HEAD_DIM + kt * 32 + hi * 16;
    aq[kt].i[0] = *(const v4i*)p;
    aq[kt].i[1] = *(const v4i*)(p + 8);
  }

  // ---- Phase 1: S = (q @ k^T) * scale over this wave's 256 columns --------
  v8f sf[16];
  const int cw = wv * 256;
#pragma unroll
  for (int t = 0; t < 16; ++t) {
    v8f c = {};
    int c0 = cw + t * 16;
#pragma unroll
    for (int kt = 0; kt < 2; ++kt) {
      // B[d][j] = k[j][d]: row-major k gives contiguous d per lane
      const _Float16* bp = kbase + (size_t)(c0 + rlo) * HEAD_DIM + kt * 32 + hi * 16;
      HFrag bf;
      bf.i[0] = *(const v4i*)bp;
      bf.i[1] = *(const v4i*)(bp + 8);
      c = __builtin_amdgcn_wmma_f32_16x16x32_f16(false, aq[kt].h, false, bf.h,
                                                 (short)0, c, false, false);
    }
#pragma unroll
    for (int j = 0; j < 8; ++j) c[j] *= 0.125f;  // HEAD_DIM^-0.5
    sf[t] = c;
  }

  // ---- Row max (wave-local, then cross-wave via LDS) ----------------------
  float m[8];
#pragma unroll
  for (int j = 0; j < 8; ++j) {
    float mm = sf[0][j];
#pragma unroll
    for (int t = 1; t < 16; ++t) mm = fmaxf(mm, sf[t][j]);
#pragma unroll
    for (int d = 1; d < 16; d <<= 1) mm = fmaxf(mm, __shfl_xor(mm, d, 32));
    m[j] = mm;
  }
  if (rlo == 0) {
#pragma unroll
    for (int j = 0; j < 8; ++j) ldsStat[wv][j + 8 * hi] = m[j];
  }
  __syncthreads();
  if (threadIdx.x < 16) {
    float mm = ldsStat[0][threadIdx.x];
    for (int w2 = 1; w2 < 8; ++w2) mm = fmaxf(mm, ldsStat[w2][threadIdx.x]);
    ldsM[threadIdx.x] = mm;
  }
  __syncthreads();
  float Mg[8];
#pragma unroll
  for (int j = 0; j < 8; ++j) Mg[j] = ldsM[j + 8 * hi];

  // ---- exp + row sum ------------------------------------------------------
#pragma unroll
  for (int t = 0; t < 16; ++t)
#pragma unroll
    for (int j = 0; j < 8; ++j) sf[t][j] = __expf(sf[t][j] - Mg[j]);
  float l[8];
#pragma unroll
  for (int j = 0; j < 8; ++j) {
    float ss = 0.f;
#pragma unroll
    for (int t = 0; t < 16; ++t) ss += sf[t][j];
#pragma unroll
    for (int d = 1; d < 16; d <<= 1) ss += __shfl_xor(ss, d, 32);
    l[j] = ss;
  }
  __syncthreads();  // all reads of ldsStat (max pass) done before reuse
  if (rlo == 0) {
#pragma unroll
    for (int j = 0; j < 8; ++j) ldsStat[wv][j + 8 * hi] = l[j];
  }
  __syncthreads();
  if (threadIdx.x < 16) {
    float ss = 0.f;
    for (int w2 = 0; w2 < 8; ++w2) ss += ldsStat[w2][threadIdx.x];
    ldsL[threadIdx.x] = ss;
  }

  // ---- Phase 2: O_partial = P_strip @ V_strip (via LDS transpose) ---------
  v8f oacc[4] = {};
  for (int it = 0; it < 8; ++it) {
    int c0 = cw + it * 32;
    // write P (2 tiles = 32 cols) into this wave's LDS strip (C layout)
#pragma unroll
    for (int tt = 0; tt < 2; ++tt) {
      int t = it * 2 + tt;
#pragma unroll
      for (int j = 0; j < 8; ++j)
        ldsP[wv][j + 8 * hi][tt * 16 + rlo] = (_Float16)sf[t][j];
    }
    __syncthreads();
    // read back in A-fragment layout
    HFrag pa;
    const _Float16* pp = &ldsP[wv][rlo][hi * 16];
    pa.i[0] = *(const v4i*)pp;
    pa.i[1] = *(const v4i*)(pp + 8);
#pragma unroll
    for (int nt = 0; nt < 4; ++nt) {
      // B[jtok][d] operand from VT: contiguous tokens per lane
      const _Float16* bp = vbase + (size_t)(nt * 16 + rlo) * SEQ + c0 + hi * 16;
      HFrag bf;
      bf.i[0] = *(const v4i*)bp;
      bf.i[1] = *(const v4i*)(bp + 8);
      oacc[nt] = __builtin_amdgcn_wmma_f32_16x16x32_f16(false, pa.h, false, bf.h,
                                                        (short)0, oacc[nt], false, false);
    }
    __syncthreads();
  }

  // ---- combine wave partials, divide by row sum, store o ------------------
#pragma unroll
  for (int nt = 0; nt < 4; ++nt)
#pragma unroll
    for (int j = 0; j < 8; ++j)
      ldsO[wv][j + 8 * hi][nt * 16 + rlo] = oacc[nt][j];
  __syncthreads();

  const int b = bh >> 4, h = bh & 15;
  for (int e = threadIdx.x; e < 16 * 64; e += 256) {
    int row = e >> 6, col = e & 63;
    float s = 0.f;
#pragma unroll
    for (int w2 = 0; w2 < 8; ++w2) s += ldsO[w2][row][col];
    float ov = s / ldsL[row];
    O[((size_t)b * SEQ + qblk * 16 + row) * (N_HEADS * HEAD_DIM) + h * HEAD_DIM + col] =
        (_Float16)ov;
  }
}

// ---------------------------------------------------------------------------
extern "C" void kernel_launch(void* const* d_in, const int* in_sizes, int n_in,
                              void* d_out, int out_size, void* d_ws, size_t ws_size,
                              hipStream_t stream) {
  (void)in_sizes; (void)n_in; (void)out_size; (void)ws_size;
  const float* src   = (const float*)d_in[0];
  const float* pnw   = (const float*)d_in[1];
  const float* pnb   = (const float*)d_in[2];
  const float* wq    = (const float*)d_in[3];
  const float* wk    = (const float*)d_in[4];
  const float* wvp   = (const float*)d_in[5];
  const float* wproj = (const float*)d_in[6];
  const float* bproj = (const float*)d_in[7];
  const float* n1w   = (const float*)d_in[8];
  const float* n1b   = (const float*)d_in[9];
  const float* w1    = (const float*)d_in[10];
  const float* b1    = (const float*)d_in[11];
  const float* w2    = (const float*)d_in[12];
  const float* b2    = (const float*)d_in[13];
  float* out = (float*)d_out;

  char* ws = (char*)d_ws;
  size_t off = 0;
  auto alloc = [&](size_t bytes) -> void* {
    void* p = (void*)(ws + off);
    off += (bytes + 255) & ~(size_t)255;
    return p;
  };
  _Float16* xh     = (_Float16*)alloc((size_t)ROWS * D_MODEL * 2);      // 16 MB
  _Float16* wqT    = (_Float16*)alloc((size_t)D_MODEL * D_MODEL * 2);   //  2 MB
  _Float16* wkT    = (_Float16*)alloc((size_t)D_MODEL * D_MODEL * 2);
  _Float16* wvT    = (_Float16*)alloc((size_t)D_MODEL * D_MODEL * 2);
  _Float16* wprojT = (_Float16*)alloc((size_t)D_MODEL * D_MODEL * 2);
  _Float16* w1T    = (_Float16*)alloc((size_t)D_FF * D_MODEL * 2);      //  8 MB
  _Float16* w2T    = (_Float16*)alloc((size_t)D_MODEL * D_FF * 2);      //  8 MB
  _Float16* qh     = (_Float16*)alloc((size_t)ROWS * D_MODEL * 2);      // 16 MB [B,H,seq,64]
  _Float16* kh     = (_Float16*)alloc((size_t)ROWS * D_MODEL * 2);      // 16 MB [B,H,seq,64]
  _Float16* vTh    = (_Float16*)alloc((size_t)ROWS * D_MODEL * 2);      // 16 MB [B,H,64,seq]
  _Float16* oh     = (_Float16*)alloc((size_t)ROWS * D_MODEL * 2);      // 16 MB
  float*    yf     = (float*)alloc((size_t)ROWS * D_MODEL * 4);         // 32 MB
  float*    s2f    = (float*)alloc((size_t)ROWS * D_MODEL * 4);         // 32 MB
  _Float16* s2h    = (_Float16*)alloc((size_t)ROWS * D_MODEL * 2);      // 16 MB
  _Float16* ff1h   = (_Float16*)alloc((size_t)ROWS * D_FF * 2);         // 64 MB

  dim3 blk(256);

  // 1) weight prep (f32 -> f16, transposed)
  transpose_to_f16<<<dim3(D_MODEL / 32, D_MODEL / 32), blk, 0, stream>>>(wq, wqT, D_MODEL, D_MODEL);
  transpose_to_f16<<<dim3(D_MODEL / 32, D_MODEL / 32), blk, 0, stream>>>(wk, wkT, D_MODEL, D_MODEL);
  transpose_to_f16<<<dim3(D_MODEL / 32, D_MODEL / 32), blk, 0, stream>>>(wvp, wvT, D_MODEL, D_MODEL);
  transpose_to_f16<<<dim3(D_MODEL / 32, D_MODEL / 32), blk, 0, stream>>>(wproj, wprojT, D_MODEL, D_MODEL);
  transpose_to_f16<<<dim3(D_MODEL / 32, D_FF / 32),    blk, 0, stream>>>(w1, w1T, D_MODEL, D_FF);
  transpose_to_f16<<<dim3(D_FF / 32,   D_MODEL / 32),  blk, 0, stream>>>(w2, w2T, D_FF, D_MODEL);

  // 2) pre-norm
  ln_novar_f16<<<ROWS, blk, 0, stream>>>(src, pnw, pnb, xh);

  // 3) Q/K/V projections (head-split / transposed-V epilogues)
  gemm_wmma_f16<<<dim3(ROWS / 256, D_MODEL / 64), blk, 0, stream>>>(
      xh, wqT, nullptr, nullptr, qh, ROWS, D_MODEL, D_MODEL, EPI_QK);
  gemm_wmma_f16<<<dim3(ROWS / 256, D_MODEL / 64), blk, 0, stream>>>(
      xh, wkT, nullptr, nullptr, kh, ROWS, D_MODEL, D_MODEL, EPI_QK);
  gemm_wmma_f16<<<dim3(ROWS / 256, D_MODEL / 64), blk, 0, stream>>>(
      xh, wvT, nullptr, nullptr, vTh, ROWS, D_MODEL, D_MODEL, EPI_VT);

  // 4) fused attention (softmax in f32, GEMMs in WMMA f16)
  attention_wmma<<<BATCH * N_HEADS * (SEQ / 16), blk, 0, stream>>>(qh, kh, vTh, oh);

  // 5) output projection + bias + residual(src) -> y (f32)
  gemm_wmma_f16<<<dim3(ROWS / 256, D_MODEL / 64), blk, 0, stream>>>(
      oh, wprojT, bproj, src, yf, ROWS, D_MODEL, D_MODEL, EPI_F_BIAS_RES);

  // 6) norm1 (dual f32 + f16 outputs)
  ln_novar_dual<<<ROWS, blk, 0, stream>>>(yf, n1w, n1b, s2f, s2h);

  // 7) FFN1: relu(s2 @ w1 + b1) -> f16
  gemm_wmma_f16<<<dim3(ROWS / 256, D_FF / 64), blk, 0, stream>>>(
      s2h, w1T, b1, nullptr, ff1h, ROWS, D_FF, D_MODEL, EPI_H_BIAS_RELU);

  // 8) FFN2 + bias + residual(s2) -> final f32 output
  gemm_wmma_f16<<<dim3(ROWS / 256, D_MODEL / 64), blk, 0, stream>>>(
      ff1h, w2T, b2, s2f, out, ROWS, D_MODEL, D_FF, EPI_F_BIAS_RES);
}
